// GCN_2327872274874
// MI455X (gfx1250) — matrix operand order repaired and checked by reading, not compiled, not run on hardware
//
#include <hip/hip_runtime.h>

// Problem constants (match reference)
#define NNODES 16384   // N
#define CHID   256     // C
#define KCLS   16      // K (classes)
#define NEDGE  524288  // E
#define KDIM   16384   // inner dim of x @ W1 (== N)

typedef __attribute__((ext_vector_type(16))) __bf16 bf16x16;
typedef __attribute__((ext_vector_type(8)))  __bf16 bf16x8;
typedef __attribute__((ext_vector_type(8)))  float  f32x8;
typedef __attribute__((ext_vector_type(4)))  float  f32x4;
typedef __attribute__((ext_vector_type(4)))  unsigned int u32x4;

#if defined(__has_builtin)
#  if __has_builtin(__builtin_amdgcn_global_load_async_to_lds_b128)
#    define HAVE_ASYNC_LDS 1
#  endif
#endif
#ifndef HAVE_ASYNC_LDS
#  define HAVE_ASYNC_LDS 0
#endif

#if HAVE_ASYNC_LDS
typedef int v4i_ __attribute__((vector_size(16)));               // matches builtin's pointee
typedef __attribute__((address_space(1))) v4i_ as1v4;            // global
typedef __attribute__((address_space(3))) v4i_ as3v4;            // LDS
#endif

__device__ __forceinline__ void wait_async0() {
#if defined(__has_builtin) && __has_builtin(__builtin_amdgcn_s_wait_asynccnt)
    __builtin_amdgcn_s_wait_asynccnt(0);
#else
    asm volatile("s_wait_asynccnt 0" ::: "memory");
#endif
}

__device__ __forceinline__ unsigned short f2bf(float f) {
    unsigned int u = __float_as_uint(f);
    u += 0x7FFFu + ((u >> 16) & 1u);   // round-to-nearest-even
    return (unsigned short)(u >> 16);
}

__device__ __forceinline__ bf16x16 cat8(bf16x8 lo, bf16x8 hi) {
    return __builtin_shufflevector(lo, hi, 0,1,2,3,4,5,6,7,8,9,10,11,12,13,14,15);
}

// ---------------------------------------------------------------- degree / dinv
__global__ __launch_bounds__(256) void deg_init_k(float* deg) {
    int i = blockIdx.x * 256 + threadIdx.x;
    deg[i] = 1.0f;                      // self-loop weight
}

__global__ __launch_bounds__(256) void deg_accum_k(const long long* __restrict__ ei,
                                                   const float* __restrict__ ew,
                                                   float* deg) {
    int e = blockIdx.x * 256 + threadIdx.x;
    int d = (int)ei[NEDGE + e];
    atomicAdd(&deg[d], ew[e]);
}

__global__ __launch_bounds__(256) void dinv_k(const float* __restrict__ deg, float* dinv) {
    int i = blockIdx.x * 256 + threadIdx.x;
    float dg = deg[i];
    dinv[i] = dg > 0.0f ? rsqrtf(fmaxf(dg, 1e-12f)) : 0.0f;
}

// ------------------------------------------------- weight panels: fp32 -> bf16, transposed
__global__ __launch_bounds__(256) void cvt_w1_k(const float* __restrict__ W1,
                                                unsigned short* __restrict__ w1bT) {
    int t = blockIdx.x * 256 + threadIdx.x;     // over CHID*KDIM
    int c = t >> 14;                            // /KDIM
    int k = t & (KDIM - 1);
    w1bT[(size_t)c * KDIM + k] = f2bf(W1[(size_t)k * CHID + c]);
}

__global__ __launch_bounds__(256) void cvt_w2_k(const float* __restrict__ W2,
                                                unsigned short* __restrict__ w2bT) {
    int t = blockIdx.x * 256 + threadIdx.x;     // over KCLS*CHID = 4096
    int c = t >> 8;
    int k = t & (CHID - 1);
    w2bT[(size_t)c * CHID + k] = f2bf(W2[(size_t)k * KCLS + c]);
}

// ---------------------------------------------------------------- GEMM1: h = x @ W1
// BM=128 (grid.x=128), BN=256 (all columns -> x streamed exactly once), BK=32.
// 8 waves: wm = wave&3 (M offset *32), wn = wave>>2 (N offset *128).
// Both tiles double-buffered in LDS, staged one K-step ahead. B tile is copied
// with GLOBAL_LOAD_ASYNC_TO_LDS_B128 (ASYNCcnt); A tile needs fp32->bf16
// conversion so it goes through VGPRs. Fragments are ds_load_b128, with the
// B-fragment stream software-pipelined one tile ahead of the wmma pair.
#define ASTRIDE 40   // LDS row stride (ushorts): 20 banks -> conflict-free, 16B aligned
#define BSTRIDE 40
__global__ __launch_bounds__(256) void gemm1_k(const float* __restrict__ x,
                                               const unsigned short* __restrict__ w1bT,
                                               float* __restrict__ h) {
    __shared__ unsigned short As[2][128 * ASTRIDE];   // 2 x 10 KB
    __shared__ unsigned short Bs[2][256 * BSTRIDE];   // 2 x 20 KB
    const int tid  = threadIdx.x;
    const int lane = tid & 31, wave = tid >> 5;
    const int wm = wave & 3, wn = wave >> 2;
    const int l16 = lane & 15, lhi = lane >> 4;
    const int rowBase = blockIdx.x * 128;

    f32x8 acc[2][8];
    #pragma unroll
    for (int i = 0; i < 2; i++)
        #pragma unroll
        for (int j = 0; j < 8; j++) acc[i][j] = f32x8{};

    // Stage one 128x32 fp32 tile of x -> bf16 into As[buf]
    auto stageA = [&](int kk, int buf) {
        #pragma unroll
        for (int i = 0; i < 4; i++) {
            int lin = tid + i * 256;
            int row = lin >> 3;
            int kq  = lin & 7;
            const float* gp = x + (size_t)(rowBase + row) * KDIM + kk + kq * 4;
            f32x4 v = *(const f32x4*)gp;
            unsigned long long packed =
                 (unsigned long long)f2bf(v.x)        |
                ((unsigned long long)f2bf(v.y) << 16) |
                ((unsigned long long)f2bf(v.z) << 32) |
                ((unsigned long long)f2bf(v.w) << 48);
            *(unsigned long long*)(&As[buf][0] + row * ASTRIDE + kq * 4) = packed;
            if (kk + 32 < KDIM) __builtin_prefetch(gp + 32, 0, 1);
        }
    };

    // Stage the 256x32 bf16 B tile: thread t copies panel row n=t (64 bytes)
    auto stageB = [&](int kk, int buf) {
        const unsigned short* gsrc = w1bT + (size_t)tid * KDIM + kk;
        unsigned short*       ldst = &Bs[buf][0] + tid * BSTRIDE;
#if HAVE_ASYNC_LDS
        #pragma unroll
        for (int ch = 0; ch < 4; ch++)
            __builtin_amdgcn_global_load_async_to_lds_b128(
                (as1v4*)(gsrc + ch * 8),
                (as3v4*)(ldst + ch * 8),
                0, 0);
#else
        #pragma unroll
        for (int ch = 0; ch < 4; ch++) {
            u32x4 v = *(const u32x4*)(gsrc + ch * 8);
            *(u32x4*)(ldst + ch * 8) = v;
        }
#endif
    };

    stageB(0, 0);
    stageA(0, 0);
    wait_async0();
    __syncthreads();

    for (int kk = 0; kk < KDIM; kk += 32) {
        const int buf = (kk >> 5) & 1;

        // Stage next K-step's tiles into the other buffers (latency hidden by wmmas)
        if (kk + 32 < KDIM) {
            stageB(kk + 32, buf ^ 1);
            stageA(kk + 32, buf ^ 1);
        }

        // A fragments: row = lane%16; lanes>=16 hold K+16..K+31 (contiguous, matches B)
        bf16x16 a[2];
        #pragma unroll
        for (int mt = 0; mt < 2; mt++) {
            int row = wm * 32 + mt * 16 + l16;
            int k0  = lhi * 16;
            bf16x8 lo = *(const bf16x8*)(&As[buf][0] + row * ASTRIDE + k0);
            bf16x8 hi = *(const bf16x8*)(&As[buf][0] + row * ASTRIDE + k0 + 8);
            a[mt] = cat8(lo, hi);
        }

        // B fragments from LDS, software-pipelined: load b(nt+1) before wmmas of b(nt)
        const unsigned short* bbase = &Bs[buf][0] + lhi * 16;
        auto loadB = [&](int nt) -> bf16x16 {
            int n = wn * 128 + nt * 16 + l16;
            return *(const bf16x16*)(bbase + n * BSTRIDE);
        };
        bf16x16 bcur = loadB(0);
        #pragma unroll
        for (int nt = 0; nt < 8; nt++) {
            bf16x16 bnxt = bcur;
            if (nt < 7) bnxt = loadB(nt + 1);   // in flight while wmmas run
            acc[0][nt] = __builtin_amdgcn_wmma_f32_16x16x32_bf16(
                false, a[0], false, bcur, (short)0, acc[0][nt], false, false);
            acc[1][nt] = __builtin_amdgcn_wmma_f32_16x16x32_bf16(
                false, a[1], false, bcur, (short)0, acc[1][nt], false, false);
            bcur = bnxt;
        }

        wait_async0();     // this wave's async B copies (buf^1) landed
        __syncthreads();   // publish both staged tiles; reads of [buf] complete
    }

    // C/D layout: N = lane%16, M = vgpr + 8*(lane/16)
    #pragma unroll
    for (int mt = 0; mt < 2; mt++) {
        int m0 = rowBase + wm * 32 + mt * 16 + lhi * 8;
        #pragma unroll
        for (int nt = 0; nt < 8; nt++) {
            int n = wn * 128 + nt * 16 + l16;
            #pragma unroll
            for (int r = 0; r < 8; r++)
                h[(size_t)(m0 + r) * CHID + n] = acc[mt][nt][r];
        }
    }
}

// ---------------------------------------------------------------- aggregation 1
__global__ __launch_bounds__(256) void agg1_init_k(const float* __restrict__ dinv,
                                                   const float* __restrict__ h,
                                                   float* __restrict__ agg1) {
    int t = blockIdx.x * 256 + threadIdx.x;     // over NNODES*CHID
    int n = t >> 8;
    float di = dinv[n];
    agg1[t] = di * di * h[t];                   // self-loop term
}

// one wave per edge; 256 channels, 8 coalesced atomics per lane
__global__ __launch_bounds__(256) void scatter1_k(const long long* __restrict__ ei,
                                                  const float* __restrict__ ew,
                                                  const float* __restrict__ dinv,
                                                  const float* __restrict__ h,
                                                  float* __restrict__ agg1) {
    int wv   = (blockIdx.x * 256 + threadIdx.x) >> 5;
    int lane = threadIdx.x & 31;
    int s = (int)ei[wv];
    int d = (int)ei[NEDGE + wv];
    float coef = dinv[s] * ew[wv] * dinv[d];
    const float* hs = h    + (size_t)s * CHID;
    float*       od = agg1 + (size_t)d * CHID;
    #pragma unroll
    for (int j = 0; j < CHID; j += 32)
        atomicAdd(&od[j + lane], coef * hs[j + lane]);
}

__global__ __launch_bounds__(256) void bias_relu_k(const float* __restrict__ agg1,
                                                   const float* __restrict__ b1,
                                                   unsigned short* __restrict__ h1b) {
    int t = blockIdx.x * 256 + threadIdx.x;
    int c = t & (CHID - 1);
    h1b[t] = f2bf(fmaxf(agg1[t] + b1[c], 0.0f));
}

// ---------------------------------------------------------------- GEMM2: h2 = h1 @ W2
__global__ __launch_bounds__(256) void gemm2_k(const unsigned short* __restrict__ h1b,
                                               const unsigned short* __restrict__ w2bT,
                                               float* __restrict__ h2) {
    const int lane = threadIdx.x & 31, wave = threadIdx.x >> 5;
    const int l16 = lane & 15, lhi = lane >> 4;
    const int mBase = (blockIdx.x * 8 + wave) * 16;

    f32x8 acc = f32x8{};
    const unsigned short* arow = h1b + (size_t)(mBase + l16) * CHID;
    const unsigned short* bcol = w2bT + (size_t)l16 * CHID;

    #pragma unroll
    for (int kk = 0; kk < CHID; kk += 32) {
        int k0a = kk + lhi * 16;
        bf16x8 lo = *(const bf16x8*)(arow + k0a);
        bf16x8 hi = *(const bf16x8*)(arow + k0a + 8);
        bf16x16 a = cat8(lo, hi);
        bf16x16 b = *(const bf16x16*)(bcol + kk + lhi * 16);
        acc = __builtin_amdgcn_wmma_f32_16x16x32_bf16(
            false, a, false, b, (short)0, acc, false, false);
    }

    int m0 = mBase + lhi * 8;
    #pragma unroll
    for (int r = 0; r < 8; r++)
        h2[(size_t)(m0 + r) * KCLS + l16] = acc[r];
}

// ---------------------------------------------------------------- aggregation 2
__global__ __launch_bounds__(256) void out_init_k(const float* __restrict__ dinv,
                                                  const float* __restrict__ h2,
                                                  const float* __restrict__ b2,
                                                  float* __restrict__ out) {
    int t = blockIdx.x * 256 + threadIdx.x;     // over NNODES*KCLS
    int n = t >> 4;
    int c = t & (KCLS - 1);
    float di = dinv[n];
    out[t] = di * di * h2[t] + b2[c];
}

__global__ __launch_bounds__(256) void scatter2_k(const long long* __restrict__ ei,
                                                  const float* __restrict__ ew,
                                                  const float* __restrict__ dinv,
                                                  const float* __restrict__ h2,
                                                  float* __restrict__ out) {
    int t = blockIdx.x * 256 + threadIdx.x;     // over NEDGE*KCLS
    int e = t >> 4;
    int c = t & (KCLS - 1);
    int s = (int)ei[e];
    int d = (int)ei[NEDGE + e];
    float coef = dinv[s] * ew[e] * dinv[d];
    atomicAdd(&out[(size_t)d * KCLS + c], coef * h2[(size_t)s * KCLS + c]);
}

// ---------------------------------------------------------------- launch
extern "C" void kernel_launch(void* const* d_in, const int* in_sizes, int n_in,
                              void* d_out, int out_size, void* d_ws, size_t ws_size,
                              hipStream_t stream) {
    const float*      x   = (const float*)d_in[0];
    const long long*  ei  = (const long long*)d_in[1];   // int64 [2][E]
    const float*      ew  = (const float*)d_in[2];
    const float*      W1  = (const float*)d_in[3];
    const float*      b1  = (const float*)d_in[4];
    const float*      W2  = (const float*)d_in[5];
    const float*      b2  = (const float*)d_in[6];
    float*            out = (float*)d_out;

    // Carve workspace (256B-aligned)
    char* ws = (char*)d_ws;
    size_t off = 0;
    auto carve = [&](size_t bytes) -> void* {
        void* p = ws + off;
        off += (bytes + 255) & ~(size_t)255;
        return p;
    };
    float*          deg  = (float*)carve((size_t)NNODES * 4);
    float*          dinv = (float*)carve((size_t)NNODES * 4);
    unsigned short* w1bT = (unsigned short*)carve((size_t)CHID * KDIM * 2);   // 8 MB
    unsigned short* w2bT = (unsigned short*)carve((size_t)KCLS * CHID * 2);
    float*          h    = (float*)carve((size_t)NNODES * CHID * 4);          // 16 MB
    float*          agg1 = (float*)carve((size_t)NNODES * CHID * 4);          // 16 MB
    float*          h2   = (float*)carve((size_t)NNODES * KCLS * 4);          // 1 MB
    unsigned short* h1b  = (unsigned short*)h;  // alias: h dead after scatter1

    deg_init_k <<<NNODES / 256, 256, 0, stream>>>(deg);
    deg_accum_k<<<NEDGE / 256, 256, 0, stream>>>(ei, ew, deg);
    dinv_k     <<<NNODES / 256, 256, 0, stream>>>(deg, dinv);

    cvt_w1_k<<<(CHID * KDIM) / 256, 256, 0, stream>>>(W1, w1bT);
    cvt_w2_k<<<(KCLS * CHID) / 256, 256, 0, stream>>>(W2, w2bT);

    gemm1_k<<<NNODES / 128, 256, 0, stream>>>(x, w1bT, h);

    agg1_init_k<<<(NNODES * CHID) / 256, 256, 0, stream>>>(dinv, h, agg1);
    scatter1_k <<<(NEDGE * 32) / 256, 256, 0, stream>>>(ei, ew, dinv, h, agg1);
    bias_relu_k<<<(NNODES * CHID) / 256, 256, 0, stream>>>(agg1, b1, h1b);

    gemm2_k<<<NNODES / (8 * 16), 256, 0, stream>>>(h1b, w2bT, h2);

    out_init_k<<<(NNODES * KCLS) / 256, 256, 0, stream>>>(dinv, h2, b2, out);
    scatter2_k<<<(NEDGE * KCLS) / 256, 256, 0, stream>>>(ei, ew, dinv, h2, out);
}
